// GRUPredictor_83416854823632
// MI455X (gfx1250) — matrix operand order repaired and checked by reading, not compiled, not run on hardware
//
#include <hip/hip_runtime.h>

// Problem constants (from reference): B=4096, T=128, D=64, ACT=2, H=64
#define Bsz  4096
#define Tt   128
#define Dd   64
#define ACTn 2
#define Hh   64
#define G3   192   // 3*H
#define XW   66    // D+ACT  (w_ih row length)

typedef _Float16 h16 __attribute__((ext_vector_type(16)));
typedef float    f8  __attribute__((ext_vector_type(8)));
typedef float    f2  __attribute__((ext_vector_type(2)));

// d_ws layout (float offsets):
//   [0,      16384)  Wf[256][64]  fused gate weights:
//                      rows   0..127 : (W1s@W_out)[j] + W_hh[j]      (r,z fused)
//                      rows 128..191 : (W1s@W_out)[j]                (i_n)
//                      rows 192..255 : W_hh[128+j']                  (h_n)
//   [16384, 16576)  beff[192] = b_ih + W1s@b_out (+ b_hh for rows<128)
//   [16640, 803072) gi0[4096][192] = s0@W1s^T + b_ih (+ b_hh rows<128)
#define WS_WF   0
#define WS_BEFF 16384
#define WS_GI0  16640

// ---------------- precompute kernels ----------------

__global__ void prep_wf(const float* __restrict__ w_ih, const float* __restrict__ w_hh,
                        const float* __restrict__ b_ih, const float* __restrict__ b_hh,
                        const float* __restrict__ b_out, const float* __restrict__ w_out,
                        float* __restrict__ ws) {
  int idx = blockIdx.x * blockDim.x + threadIdx.x;   // 0 .. 256*64
  if (idx < 256 * Hh) {
    int j = idx / Hh, k = idx % Hh;
    float v;
    if (j < G3) {
      // Wc[j][k] = sum_d W1s[j][d] * w_out[d][k]
      float acc = 0.f;
      for (int d = 0; d < Dd; ++d) acc += w_ih[j * XW + d] * w_out[d * Hh + k];
      v = acc;
      if (j < 2 * Hh) v += w_hh[j * Hh + k];         // fuse r,z hidden weights
    } else {
      v = w_hh[(j - Hh) * Hh + k];                   // h_n rows = W_hh rows 128..191
    }
    ws[WS_WF + j * Hh + k] = v;
  }
  if (idx < G3) {
    int j = idx;
    float bc = 0.f;
    for (int d = 0; d < Dd; ++d) bc += w_ih[j * XW + d] * b_out[d];
    float v = b_ih[j] + bc;
    if (j < 2 * Hh) v += b_hh[j];
    ws[WS_BEFF + j] = v;
  }
}

__global__ void prep_gi0(const float* __restrict__ s0, const float* __restrict__ w_ih,
                         const float* __restrict__ b_ih, const float* __restrict__ b_hh,
                         float* __restrict__ ws) {
  int idx = blockIdx.x * blockDim.x + threadIdx.x;   // B*192
  if (idx >= Bsz * G3) return;
  int b = idx / G3, j = idx % G3;
  float acc = b_ih[j] + ((j < 2 * Hh) ? b_hh[j] : 0.f);
  const float* srow = s0 + (size_t)b * Dd;
  const float* wrow = w_ih + (size_t)j * XW;
  for (int d = 0; d < Dd; ++d) acc += srow[d] * wrow[d];
  ws[WS_GI0 + idx] = acc;
}

// ---------------- gate nonlinearities ----------------

#if __has_builtin(__builtin_amdgcn_tanhf)
__device__ __forceinline__ float tanh_f(float x) { return __builtin_amdgcn_tanhf(x); }
__device__ __forceinline__ float sig_f(float x)  { return 0.5f + 0.5f * __builtin_amdgcn_tanhf(0.5f * x); }
#else
__device__ __forceinline__ float sig_f(float x)  { return __builtin_amdgcn_rcpf(1.0f + __expf(-x)); }
__device__ __forceinline__ float tanh_f(float x) { return 2.0f * sig_f(2.0f * x) - 1.0f; }
#endif

// ---------------- main recurrence kernel ----------------
// 2 waves (64 threads) per block; the block owns 16 batch rows for all 128
// steps. Wave w owns hidden/output COLUMN half [32w, 32w+32): it computes the
// corresponding fused-gate tiles, its half of h_t (== A-fragment K-tile w),
// and its half of the output projection. One __syncthreads per step exchanges
// h through a double-buffered LDS staging area.

__global__ __launch_bounds__(64, 1) void gru_main(
    const float* __restrict__ a, const float* __restrict__ w_ih,
    const float* __restrict__ b_hh, const float* __restrict__ w_out,
    const float* __restrict__ b_out, const float* __restrict__ ws,
    float* __restrict__ out)
{
  // LDS: 32 f16 gate B-fragments, each 32 lanes * 24-half slot (48B stride,
  // 16B aligned) + double-buffered h staging (2 bufs * 2 ktiles * 768 halfs).
  __shared__ __align__(16) _Float16 smem[32 * 768 + 2 * 1536];
  _Float16* stageS = smem + 32 * 768;

  const int tid     = threadIdx.x;
  const int lane    = tid & 31;
  const int waveId  = tid >> 5;            // 0/1: column half owned
  const int lanelow = lane & 15;
  const int lanehi  = lane >> 4;
  const int mbase   = lanehi * 8;          // C-layout: lanes 16..31 hold rows M=8..15
  const int rowbase = blockIdx.x * 16;

  // ---- preload fused gate weights as f16 B-fragments into LDS ----
  // wave w loads K-tile kt == w of every N-tile.
  // B layout used here (contiguous k, same order as the A staging): lane holds col n = lane%16,
  // halfs q -> K = kt*32 + (lane<16 ? 0 : 16) + q
  const float* wf = ws + WS_WF;
#pragma unroll
  for (int nt = 0; nt < 16; ++nt) {
    const float* src = wf + (nt * 16 + lanelow) * Hh + waveId * 32 + lanehi * 16;
    _Float16* dst = smem + (nt * 2 + waveId) * 768 + lane * 24;
#pragma unroll
    for (int q = 0; q < 16; ++q) dst[q] = (_Float16)src[q];
  }

  // ---- this wave's two w_out B-fragment column tiles, in VGPRs ----
  h16 wo[2][2];
#pragma unroll
  for (int ctl = 0; ctl < 2; ++ctl) {
    int ct = 2 * waveId + ctl;
#pragma unroll
    for (int kt = 0; kt < 2; ++kt) {
      const float* src = w_out + (ct * 16 + lanelow) * Hh + kt * 32 + lanehi * 16;
      h16 v;
#pragma unroll
      for (int q = 0; q < 16; ++q) v[q] = (_Float16)src[q];
      wo[ctl][kt] = v;
    }
  }

  // ---- f32 K=4 B-fragments for accumulator init (rank-2 a-update + biases) ----
  // Assumed 32-bit 4x16 B layout (mirror of documented 16x4 A layout):
  //   reg0: lanes 0-15 K=0, lanes 16-31 K=2 ; reg1: lanes 0-15 K=1, lanes 16-31 K=3
  // K slots: K0 = a0 weight, K1 = a1 weight, K2 = beff (gated), K3 = b_hh_n.
  f2 wabR[2], wabZ[2], wabN[2], wabH[2];
#pragma unroll
  for (int ctl = 0; ctl < 2; ++ctl) {
    int jc = (2 * waveId + ctl) * 16 + lanelow;        // column feature 0..63
    wabR[ctl][0] = lanehi ? ws[WS_BEFF + jc]        : w_ih[jc * XW + 64];
    wabR[ctl][1] = lanehi ? 0.0f                    : w_ih[jc * XW + 65];
    wabZ[ctl][0] = lanehi ? ws[WS_BEFF + 64 + jc]   : w_ih[(64 + jc) * XW + 64];
    wabZ[ctl][1] = lanehi ? 0.0f                    : w_ih[(64 + jc) * XW + 65];
    wabN[ctl][0] = lanehi ? ws[WS_BEFF + 128 + jc]  : w_ih[(128 + jc) * XW + 64];
    wabN[ctl][1] = lanehi ? 0.0f                    : w_ih[(128 + jc) * XW + 65];
    wabH[ctl][0] = 0.0f;
    wabH[ctl][1] = lanehi ? b_hh[128 + jc] : 0.0f;
  }

  float bout_l[2];
#pragma unroll
  for (int ctl = 0; ctl < 2; ++ctl)
    bout_l[ctl] = b_out[(2 * waveId + ctl) * 16 + lanelow];

  // staging scatter addresses: element (M, n=(2w+ctl)*16+lanelow) ->
// kt = n>>5 == waveId, k32 = n&31, target lane = M + ((k32&16)?16:0),
// half = k32&15
  int stc[2];
#pragma unroll
  for (int ctl = 0; ctl < 2; ++ctl) {
    int n   = (2 * waveId + ctl) * 16 + lanelow;
    int k32 = n & 31;
    int hi  = (k32 >> 4) & 1;
    int hh  = k32 & 15;
    stc[ctl] = waveId * 768 + hi * 384 + mbase * 24 + hh;   // + r*24 per row
  }

  __syncthreads();

  // h state: this wave's two f32 C-layout col-tiles + full f16 A-fragments
  f8  hC[2];
  h16 ha[2];
#pragma unroll
  for (int ctl = 0; ctl < 2; ++ctl)
#pragma unroll
    for (int r = 0; r < 8; ++r) hC[ctl][r] = 0.0f;
#pragma unroll
  for (int kt = 0; kt < 2; ++kt)
#pragma unroll
    for (int q = 0; q < 16; ++q) ha[kt][q] = (_Float16)0.0f;

  f8 zero8;
#pragma unroll
  for (int r = 0; r < 8; ++r) zero8[r] = 0.0f;

  const float* gi0row = ws + WS_GI0 + (size_t)rowbase * G3;
  const float* arow   = a + (size_t)(rowbase + lanelow) * Tt * ACTn;

#pragma unroll 1
  for (int t = 0; t < Tt; ++t) {
    _Float16* sb = stageS + (t & 1) * 1536;   // double-buffered staging

    // f32 A-fragment for init-WMMA (16x4, documented layout:
    // reg0: lanes0-15 K0, lanes16-31 K2 ; reg1: lanes0-15 K1, lanes16-31 K3)
    const float biasEn = (t == 0) ? 0.0f : 1.0f;   // at t=0 bias lives in gi0
    float2 av = *(const float2*)(arow + (size_t)t * ACTn);
    f2 aA;
    aA[0] = lanehi ? biasEn : av.x;
    aA[1] = lanehi ? 1.0f   : av.y;

    f8 hN2[2];
#pragma unroll
    for (int ctl = 0; ctl < 2; ++ctl) {
      const int ct = 2 * waveId + ctl;       // global column tile 0..3
      f8 cR, cZ, cN;
      if (t == 0) {        // unfused path once: s0-driven pre-activations
#pragma unroll
        for (int r = 0; r < 8; ++r) {
          const float* g0 = gi0row + (size_t)(mbase + r) * G3 + ct * 16 + lanelow;
          cR[r] = g0[0];
          cZ[r] = g0[64];
          cN[r] = g0[128];
        }
      } else {
        cR = zero8; cZ = zero8; cN = zero8;
      }

      // accumulator init as rank-4 f32 WMMA: a-part + biases
      f8 aR = __builtin_amdgcn_wmma_f32_16x16x4_f32(false, aA, false, wabR[ctl], (short)0, cR,    false, false);
      f8 aZ = __builtin_amdgcn_wmma_f32_16x16x4_f32(false, aA, false, wabZ[ctl], (short)0, cZ,    false, false);
      f8 aN = __builtin_amdgcn_wmma_f32_16x16x4_f32(false, aA, false, wabN[ctl], (short)0, cN,    false, false);
      f8 aH = __builtin_amdgcn_wmma_f32_16x16x4_f32(false, aA, false, wabH[ctl], (short)0, zero8, false, false);

      // fused gate GEMM: acc += h_{t-1} @ Wf^T   (at t=0, ha==0 -> no-op)
#pragma unroll
      for (int kt = 0; kt < 2; ++kt) {
        const h16 bR = *(const h16*)(smem + ((ct)      * 2 + kt) * 768 + lane * 24);
        const h16 bZ = *(const h16*)(smem + ((4 + ct)  * 2 + kt) * 768 + lane * 24);
        const h16 bN = *(const h16*)(smem + ((8 + ct)  * 2 + kt) * 768 + lane * 24);
        const h16 bH = *(const h16*)(smem + ((12 + ct) * 2 + kt) * 768 + lane * 24);
        aR = __builtin_amdgcn_wmma_f32_16x16x32_f16(false, ha[kt], false, bR, (short)0, aR, false, false);
        aZ = __builtin_amdgcn_wmma_f32_16x16x32_f16(false, ha[kt], false, bZ, (short)0, aZ, false, false);
        aN = __builtin_amdgcn_wmma_f32_16x16x32_f16(false, ha[kt], false, bN, (short)0, aN, false, false);
        aH = __builtin_amdgcn_wmma_f32_16x16x32_f16(false, ha[kt], false, bH, (short)0, aH, false, false);
      }

      // gates: r = sig, z = sig, n = tanh(i_n + r*h_n), h' = n + z*(h-n)
#pragma unroll
      for (int r = 0; r < 8; ++r) {
        float rg = sig_f(aR[r]);
        float zg = sig_f(aZ[r]);
        float ng = tanh_f(aN[r] + rg * aH[r]);
        hN2[ctl][r] = ng + zg * (hC[ctl][r] - ng);
      }
      hC[ctl] = hN2[ctl];
    }

    // scatter this wave's f16 h columns into A-fragment staging (K-tile waveId)
#pragma unroll
    for (int ctl = 0; ctl < 2; ++ctl)
#pragma unroll
      for (int r = 0; r < 8; ++r)
        sb[stc[ctl] + r * 24] = (_Float16)hN2[ctl][r];

    __syncthreads();   // waits dscnt, then 2-wave barrier: h fully staged

    ha[0] = *(const h16*)(sb + 0 * 768 + lane * 24);
    ha[1] = *(const h16*)(sb + 1 * 768 + lane * 24);

    // output projection: this wave's columns of s_t = h_t @ w_out^T + b_out
#pragma unroll
    for (int ctl = 0; ctl < 2; ++ctl) {
      f8 sa;
#pragma unroll
      for (int r = 0; r < 8; ++r) sa[r] = bout_l[ctl];
#pragma unroll
      for (int kt = 0; kt < 2; ++kt)
        sa = __builtin_amdgcn_wmma_f32_16x16x32_f16(false, ha[kt], false, wo[ctl][kt], (short)0, sa, false, false);
#pragma unroll
      for (int r = 0; r < 8; ++r)
        out[((size_t)(rowbase + mbase + r) * Tt + t) * Dd + (2 * waveId + ctl) * 16 + lanelow] = sa[r];
    }
  }
}

// ---------------- launch ----------------

extern "C" void kernel_launch(void* const* d_in, const int* in_sizes, int n_in,
                              void* d_out, int out_size, void* d_ws, size_t ws_size,
                              hipStream_t stream) {
  (void)in_sizes; (void)n_in; (void)out_size; (void)ws_size;
  const float* s0    = (const float*)d_in[0];
  const float* a     = (const float*)d_in[1];
  const float* w_ih  = (const float*)d_in[2];
  const float* w_hh  = (const float*)d_in[3];
  const float* b_ih  = (const float*)d_in[4];
  const float* b_hh  = (const float*)d_in[5];
  const float* w_out = (const float*)d_in[6];
  const float* b_out = (const float*)d_in[7];
  float* out = (float*)d_out;
  float* ws  = (float*)d_ws;

  // fused weights + biases (16384 threads covers Wf; first 192 also do beff)
  prep_wf<<<(256 * Hh + 255) / 256, 256, 0, stream>>>(w_ih, w_hh, b_ih, b_hh, b_out, w_out, ws);
  // t=0 gate pre-activations from s0
  prep_gi0<<<(Bsz * G3 + 255) / 256, 256, 0, stream>>>(s0, w_ih, b_ih, b_hh, ws);
  // recurrence: 256 blocks of 2 waves; block owns 16 rows, wave owns 32 cols
  gru_main<<<Bsz / 16, 64, 0, stream>>>(a, w_ih, b_hh, w_out, b_out, ws, out);
}
